// Clause_GNN_81088982548905
// MI455X (gfx1250) — matrix operand order repaired and checked by reading, not compiled, not run on hardware
//
#include <hip/hip_runtime.h>
#include <math.h>

// ---------------------------------------------------------------------------
// Clause-GNN for MI455X (gfx1250): WMMA f16 GEMMs + persistent GRU scan.
// B=4, T=128, N=64, D=768, H=512, HG=256
// B-tiles staged to LDS via GLOBAL_LOAD_ASYNC_TO_LDS_B128 (ASYNCcnt path).
// ---------------------------------------------------------------------------

typedef __attribute__((ext_vector_type(16))) _Float16 v16h;
typedef __attribute__((ext_vector_type(8)))  float    v8f;

#define CB   4
#define CT   128
#define CN   64
#define CD   768
#define CH   512
#define CHG  256

// ---- WMMA fragment loaders (contiguous k = hi*16+e in A and B, wave32) ----------
__device__ __forceinline__ v16h frag_a_lds(const _Float16* A, int lda, int lane) {
  int m = lane & 15, hi = lane >> 4;
  v16h a;
#pragma unroll
  for (int e = 0; e < 16; ++e) {
    int k = hi * 16 + e;   // A 16x32 f16: k = hi*16+e (matches B)
    a[e] = A[m * lda + k];
  }
  return a;
}

__device__ __forceinline__ v16h frag_b_lds(const _Float16* Bp, int ldb, int lane) {
  int n = lane & 15, hi = lane >> 4;
  v16h b;
#pragma unroll
  for (int e = 0; e < 16; ++e) {                     // B 32x16 f16: k = hi*16+e
    b[e] = Bp[(hi * 16 + e) * ldb + n];
  }
  return b;
}

// ---------------------------------------------------------------------------
// Generic tiled WMMA GEMM: C[M,Nn] = act(A' @ Wt + bias)
//   Wt: f16, K-major [K][Nn] (pre-transposed weights)
//   fused==0: A' = A (f32 row-major [M,K])
//   fused==1: A'[m][k] = rel[m*K+k] * gate[((m>>12)*64 + (m&63))*K + k]
//   act: 0 = none, 1 = sigmoid
// Block: 256 threads = 8 waves -> 32x64 tile; grid = (M/32, Nn/64)
// B tile copied global->LDS with async-LDS b128 (no VGPR round trip).
// ---------------------------------------------------------------------------
__global__ __launch_bounds__(256)
void wmma_gemm(const float* __restrict__ A, const float* __restrict__ rel,
               const float* __restrict__ gate, const _Float16* __restrict__ Wt,
               const float* __restrict__ bias, float* __restrict__ C,
               int M, int Nn, int K, int act, int fused) {
  __shared__ _Float16 As[32 * 32];
  __shared__ _Float16 Bs[32 * 64];
  int tid = threadIdx.x;
  int wave = tid >> 5, lane = tid & 31;
  int wm = wave >> 2, wn = wave & 3;
  int m0 = blockIdx.x * 32;
  int n0 = blockIdx.y * 64;
  // per-thread B-tile copy slot: 8 halfs = 16 bytes
  int bidx = tid * 8;
  int brow = bidx >> 6, bcol = bidx & 63;
  unsigned lds_dst = (unsigned)(size_t)(&Bs[brow * 64 + bcol]);
  v8f acc = {};
  for (int k0 = 0; k0 < K; k0 += 32) {
    // async-copy B tile 32x64 f16 (contiguous K-major rows) straight to LDS
    {
      const _Float16* src = Wt + (size_t)(k0 + brow) * Nn + n0 + bcol;
      asm volatile("global_load_async_to_lds_b128 %0, %1, off"
                   :: "v"(lds_dst), "v"(src) : "memory");
    }
    // stage A tile 32x32 (f32 -> f16), 4 elems/thread
    {
      int idx = tid * 4;
      int row = idx >> 5, col = idx & 31;
      int mg = m0 + row;
      float4 va;
      if (fused) {
        float4 r4 = *(const float4*)(rel + (size_t)mg * K + k0 + col);
        int gr = ((mg >> 12) << 6) + (mg & 63);
        float4 g4 = *(const float4*)(gate + (size_t)gr * K + k0 + col);
        va.x = r4.x * g4.x; va.y = r4.y * g4.y;
        va.z = r4.z * g4.z; va.w = r4.w * g4.w;
      } else {
        va = *(const float4*)(A + (size_t)mg * K + k0 + col);
      }
      As[row * 32 + col + 0] = (_Float16)va.x;
      As[row * 32 + col + 1] = (_Float16)va.y;
      As[row * 32 + col + 2] = (_Float16)va.z;
      As[row * 32 + col + 3] = (_Float16)va.w;
    }
    if (k0 + 32 < K)  // prefetch next B tile rows
      __builtin_prefetch(Wt + (size_t)(k0 + 32 + (tid >> 3)) * Nn + n0, 0, 3);
    asm volatile("s_wait_asynccnt 0x0" ::: "memory");
    __syncthreads();
    v16h af = frag_a_lds(&As[wm * 16 * 32], 32, lane);
    v16h bf = frag_b_lds(&Bs[wn * 16], 64, lane);
    acc = __builtin_amdgcn_wmma_f32_16x16x32_f16(false, af, false, bf,
                                                 (short)0, acc, false, false);
    __syncthreads();
  }
  int nl = lane & 15, hi = lane >> 4;
  int nG = n0 + wn * 16 + nl;
  float bv = bias ? bias[nG] : 0.f;
#pragma unroll
  for (int r = 0; r < 8; ++r) {
    int mG = m0 + wm * 16 + hi * 8 + r;
    float v = acc[r] + bv;
    if (act == 1) v = 1.f / (1.f + __expf(-v));
    C[(size_t)mG * Nn + nG] = v;
  }
}

// ---------------------------------------------------------------------------
// Batched small GEMM (A f32 [64,lda] per batch, B f32 [.,512] K-major rows):
//   C[b,m, col0+n] = post(sum_k A[b,m,k] * Bsrc[(b*K+k)*512+n])
// post: divide by divv[b*64+m] (pooling) and/or relu. grid=(2, 8, B)
// ---------------------------------------------------------------------------
__global__ __launch_bounds__(256)
void av_gemm(const float* __restrict__ A, int lda, const float* __restrict__ Bsrc,
             int K, const float* __restrict__ divv, float* __restrict__ C,
             int ldc, int col0, int relu) {
  __shared__ _Float16 As[32 * 32];
  __shared__ _Float16 Bs[32 * 64];
  int b = blockIdx.z;
  int m0 = blockIdx.x * 32, n0 = blockIdx.y * 64;
  int tid = threadIdx.x, wave = tid >> 5, lane = tid & 31;
  int wm = wave >> 2, wn = wave & 3;
  const float* Ab = A + (size_t)b * 64 * lda;
  v8f acc = {};
  for (int k0 = 0; k0 < K; k0 += 32) {
    {
      int idx = tid * 4, row = idx >> 5, col = idx & 31;
      float4 va = *(const float4*)(Ab + (size_t)(m0 + row) * lda + k0 + col);
      As[row * 32 + col + 0] = (_Float16)va.x;
      As[row * 32 + col + 1] = (_Float16)va.y;
      As[row * 32 + col + 2] = (_Float16)va.z;
      As[row * 32 + col + 3] = (_Float16)va.w;
    }
    {
      int idx = tid * 8, row = idx >> 6, col = idx & 63;
      const float* bp = Bsrc + ((size_t)(b * K + k0 + row)) * 512 + n0 + col;
      float4 v0 = *(const float4*)bp;
      float4 v1 = *(const float4*)(bp + 4);
      Bs[row * 64 + col + 0] = (_Float16)v0.x;
      Bs[row * 64 + col + 1] = (_Float16)v0.y;
      Bs[row * 64 + col + 2] = (_Float16)v0.z;
      Bs[row * 64 + col + 3] = (_Float16)v0.w;
      Bs[row * 64 + col + 4] = (_Float16)v1.x;
      Bs[row * 64 + col + 5] = (_Float16)v1.y;
      Bs[row * 64 + col + 6] = (_Float16)v1.z;
      Bs[row * 64 + col + 7] = (_Float16)v1.w;
    }
    __syncthreads();
    v16h af = frag_a_lds(&As[wm * 16 * 32], 32, lane);
    v16h bf = frag_b_lds(&Bs[wn * 16], 64, lane);
    acc = __builtin_amdgcn_wmma_f32_16x16x32_f16(false, af, false, bf,
                                                 (short)0, acc, false, false);
    __syncthreads();
  }
  int nl = lane & 15, hi = lane >> 4;
  int nG = n0 + wn * 16 + nl;
#pragma unroll
  for (int r = 0; r < 8; ++r) {
    int mG = m0 + wm * 16 + hi * 8 + r;
    float v = acc[r];
    if (divv) v /= divv[b * 64 + mG];
    if (relu) v = fmaxf(v, 0.f);
    C[((size_t)(b * 64 + mG)) * ldc + col0 + nG] = v;
  }
}

// ---------------------------------------------------------------------------
// Persistent GRU scan for one direction: gi [B*T,768] (x@Wih^T+bih precomputed)
// h kept in LDS (f16 for WMMA A + f32 master). gh = h@Whh^T via WMMA.
// ---------------------------------------------------------------------------
__global__ __launch_bounds__(256)
void gru_scan(const float* __restrict__ gi, const _Float16* __restrict__ whhT,
              const float* __restrict__ bhh, float* __restrict__ y,
              int ycol0, int rev) {
  __shared__ _Float16 h16[16 * 256];   // rows 4..15 stay zero (WMMA M padding)
  __shared__ float    h32[4 * 256];
  __shared__ float    gh[4 * 768];
  int tid = threadIdx.x;
  int wave = tid >> 5, lane = tid & 31;
  for (int i = tid; i < 16 * 256; i += 256) h16[i] = (_Float16)0.f;
  for (int i = tid; i < 4 * 256; i += 256) h32[i] = 0.f;
  __syncthreads();
  for (int t = 0; t < CT; ++t) {
    int tt = rev ? (CT - 1 - t) : t;
    int hi = lane >> 4;
    // gh = h @ Whh^T : 48 N-tiles of 16, 6 per wave, K=256 in 8 steps
    for (int q = 0; q < 6; ++q) {
      int nt = q * 8 + wave;
      int ncol = nt * 16 + (lane & 15);
      v8f acc = {};
      for (int kb = 0; kb < 8; ++kb) {
        v16h af = frag_a_lds(&h16[kb * 32], 256, lane);
        v16h bf;
#pragma unroll
        for (int e = 0; e < 16; ++e)
          bf[e] = whhT[(size_t)(kb * 32 + hi * 16 + e) * 768 + ncol];
        acc = __builtin_amdgcn_wmma_f32_16x16x32_f16(false, af, false, bf,
                                                     (short)0, acc, false, false);
      }
      if (hi == 0) {
#pragma unroll
        for (int r = 0; r < 4; ++r) gh[r * 768 + ncol] = acc[r];
      }
    }
    __syncthreads();
    // gate update: 4*256 = 1024 elems, 4 per thread
#pragma unroll
    for (int i = 0; i < 4; ++i) {
      int idx = tid + i * 256;
      int b = idx >> 8, col = idx & 255;
      const float* grow = gi + (size_t)(b * CT + tt) * 768;
      float ir = grow[col]       + gh[b * 768 + col]       + bhh[col];
      float iz = grow[col + 256] + gh[b * 768 + col + 256] + bhh[col + 256];
      float hn = gh[b * 768 + col + 512] + bhh[col + 512];
      float r  = 1.f / (1.f + __expf(-ir));
      float z  = 1.f / (1.f + __expf(-iz));
      float nn = tanhf(grow[col + 512] + r * hn);
      float h2 = (1.f - z) * nn + z * h32[b * 256 + col];
      h32[b * 256 + col] = h2;
      h16[b * 256 + col] = (_Float16)h2;
      y[(size_t)(b * CT + tt) * 512 + ycol0 + col] = h2;
    }
    __syncthreads();
  }
}

// ---- LayerNorm + leaky_relu + dot(dw) per row of a[16384,512] --------------
__global__ __launch_bounds__(256)
void ln_leaky_dot(const float* __restrict__ a, const float* __restrict__ g,
                  const float* __restrict__ be, const float* __restrict__ dw,
                  const float* __restrict__ db, float* __restrict__ logits) {
  int wave = threadIdx.x >> 5, lane = threadIdx.x & 31;
  size_t row = (size_t)blockIdx.x * 8 + wave;
  const float* ar = a + row * 512;
  float x[16], s = 0.f, ss = 0.f;
#pragma unroll
  for (int i = 0; i < 16; ++i) {
    x[i] = ar[lane + i * 32];
    s += x[i]; ss += x[i] * x[i];
  }
#pragma unroll
  for (int o = 16; o > 0; o >>= 1) {
    s += __shfl_xor(s, o, 32); ss += __shfl_xor(ss, o, 32);
  }
  float mu = s * (1.f / 512.f);
  float var = ss * (1.f / 512.f) - mu * mu;
  float rs = rsqrtf(var + 1e-5f);
  float dot = 0.f;
#pragma unroll
  for (int i = 0; i < 16; ++i) {
    int e = lane + i * 32;
    float c = (x[i] - mu) * rs * g[e] + be[e];
    c = (c > 0.f) ? c : 0.01f * c;
    dot += c * dw[e];
  }
#pragma unroll
  for (int o = 16; o > 0; o >>= 1) dot += __shfl_xor(dot, o, 32);
  if (lane == 0) logits[row] = dot + db[0];
}

// ---- masked softmax over j (row of 64), att = softmax(ml)*adj --------------
__global__ __launch_bounds__(256)
void softmax_att(const float* __restrict__ logits, const float* __restrict__ adj,
                 float* __restrict__ att) {
  int wave = threadIdx.x >> 5, lane = threadIdx.x & 31;
  int row = blockIdx.x * 8 + wave;       // row = b*64 + i
  int b = row >> 6, i = row & 63;
  float ml[2], ad[2];
#pragma unroll
  for (int jj = 0; jj < 2; ++jj) {
    int j = lane + jj * 32;
    float a = adj[(size_t)b * 4096 + i * 64 + j];
    float lg = logits[(size_t)row * 64 + j] + ((i == j) ? 1.f : 0.f);
    ml[jj] = lg * a + (1.f - a) * (-1e30f);
    ad[jj] = a;
  }
  float mx = fmaxf(ml[0], ml[1]);
#pragma unroll
  for (int o = 16; o > 0; o >>= 1) mx = fmaxf(mx, __shfl_xor(mx, o, 32));
  float e0 = __expf(ml[0] - mx), e1 = __expf(ml[1] - mx);
  float s = e0 + e1;
#pragma unroll
  for (int o = 16; o > 0; o >>= 1) s += __shfl_xor(s, o, 32);
  float inv = 1.f / s;
  att[(size_t)row * 64 + lane]      = e0 * inv * ad[0];
  att[(size_t)row * 64 + lane + 32] = e1 * inv * ad[1];
}

// ---- wnt[b,n] = max(sum_t adj3[b,n,t], 1) ----------------------------------
__global__ void wnt_kernel(const float* __restrict__ adj3, float* __restrict__ wnt) {
  int idx = threadIdx.x;  // 256 = B*N
  const float* p = adj3 + (size_t)idx * CT;
  float s = 0.f;
  for (int t = 0; t < CT; ++t) s += p[t];
  wnt[idx] = (s == 0.f) ? 1.f : s;
}

// ---- Wt[k*Nn+n] = (f16) W[n*K+k]  (pre-transpose weights to K-major f16) ---
__global__ void convert_wt(const float* __restrict__ W, _Float16* __restrict__ Wt,
                           int Nn, int K) {
  size_t idx = (size_t)blockIdx.x * 256 + threadIdx.x;
  if (idx >= (size_t)Nn * K) return;
  int n = (int)(idx % Nn);
  size_t k = idx / Nn;
  Wt[idx] = (_Float16)W[(size_t)n * K + k];
}

// ---------------------------------------------------------------------------
extern "C" void kernel_launch(void* const* d_in, const int* in_sizes, int n_in,
                              void* d_out, int out_size, void* d_ws, size_t ws_size,
                              hipStream_t stream) {
  (void)in_sizes; (void)n_in; (void)out_size; (void)ws_size;
  const float* gru_input = (const float*)d_in[0];
  const float* rel_mask  = (const float*)d_in[1];
  const float* rel_adj   = (const float*)d_in[2];
  const float* adj3      = (const float*)d_in[3];
  const float *wih[2][2], *whhp[2][2], *bih[2][2], *bhhp[2][2];
  int ii = 4;
  for (int l = 0; l < 2; ++l)
    for (int d = 0; d < 2; ++d) {
      wih[l][d]  = (const float*)d_in[ii++];
      whhp[l][d] = (const float*)d_in[ii++];
      bih[l][d]  = (const float*)d_in[ii++];
      bhhp[l][d] = (const float*)d_in[ii++];
    }
  const float* f_w  = (const float*)d_in[20];
  const float* f_b  = (const float*)d_in[21];
  const float* a_w  = (const float*)d_in[22];
  const float* a_b  = (const float*)d_in[23];
  const float* ln_g = (const float*)d_in[24];
  const float* ln_b = (const float*)d_in[25];
  const float* d_wp = (const float*)d_in[26];
  const float* d_bp = (const float*)d_in[27];
  const float* merge_w = (const float*)d_in[28];
  const float* merge_b = (const float*)d_in[29];

  // workspace carve-out
  char* base = (char*)d_ws;
  size_t off = 0;
  auto alloc = [&](size_t bytes) -> void* {
    void* p = base + off;
    off = (off + bytes + 255) & ~(size_t)255;
    return p;
  };
  _Float16* wih_t[2][2]; _Float16* whh_t[2][2];
  wih_t[0][0] = (_Float16*)alloc(768 * 768 * 2);
  wih_t[0][1] = (_Float16*)alloc(768 * 768 * 2);
  wih_t[1][0] = (_Float16*)alloc(512 * 768 * 2);
  wih_t[1][1] = (_Float16*)alloc(512 * 768 * 2);
  for (int l = 0; l < 2; ++l)
    for (int d = 0; d < 2; ++d) whh_t[l][d] = (_Float16*)alloc(256 * 768 * 2);
  _Float16* fw_t[6]; _Float16* aw_t[6];
  for (int h = 0; h < 6; ++h) fw_t[h] = (_Float16*)alloc(512 * 768 * 2);
  for (int h = 0; h < 6; ++h) aw_t[h] = (_Float16*)alloc(768 * 512 * 2);
  _Float16* merge_t = (_Float16*)alloc((size_t)3072 * 256 * 2);
  float* gi[2];
  gi[0] = (float*)alloc((size_t)512 * 768 * 4);
  gi[1] = (float*)alloc((size_t)512 * 768 * 4);
  float* x1      = (float*)alloc((size_t)512 * 512 * 4);
  float* gru_out = (float*)alloc((size_t)512 * 512 * 4);
  float* wnt     = (float*)alloc(256 * 4);
  float* feat0   = (float*)alloc((size_t)256 * 512 * 4);
  float* featw   = (float*)alloc((size_t)256 * 512 * 4);
  float* gateb   = (float*)alloc((size_t)256 * 768 * 4);
  float* abuf    = (float*)alloc((size_t)16384 * 512 * 4);
  float* logits  = (float*)alloc((size_t)16384 * 4);
  float* att     = (float*)alloc((size_t)16384 * 4);
  float* concat  = (float*)alloc((size_t)256 * 3072 * 4);

  auto cvt = [&](const float* W, _Float16* Wt, int Nn, int K) {
    int total = Nn * K;
    convert_wt<<<(total + 255) / 256, 256, 0, stream>>>(W, Wt, Nn, K);
  };
  // weight conversion (one-time per launch)
  cvt(wih[0][0], wih_t[0][0], 768, 768);
  cvt(wih[0][1], wih_t[0][1], 768, 768);
  cvt(wih[1][0], wih_t[1][0], 768, 512);
  cvt(wih[1][1], wih_t[1][1], 768, 512);
  for (int l = 0; l < 2; ++l)
    for (int d = 0; d < 2; ++d) cvt(whhp[l][d], whh_t[l][d], 768, 256);
  for (int h = 0; h < 6; ++h) cvt(f_w + (size_t)h * 768 * 512, fw_t[h], 768, 512);
  for (int h = 0; h < 6; ++h) cvt(a_w + (size_t)h * 512 * 768, aw_t[h], 512, 768);
  cvt(merge_w, merge_t, 256, 3072);

  // GRU layer 0: input projections then scans (fwd into cols 0..255, bwd 256..511)
  for (int d = 0; d < 2; ++d)
    wmma_gemm<<<dim3(16, 12), 256, 0, stream>>>(gru_input, nullptr, nullptr,
        wih_t[0][d], bih[0][d], gi[d], 512, 768, 768, 0, 0);
  for (int d = 0; d < 2; ++d)
    gru_scan<<<1, 256, 0, stream>>>(gi[d], whh_t[0][d], bhhp[0][d], x1, d * 256, d);
  // GRU layer 1
  for (int d = 0; d < 2; ++d)
    wmma_gemm<<<dim3(16, 12), 256, 0, stream>>>(x1, nullptr, nullptr,
        wih_t[1][d], bih[1][d], gi[d], 512, 768, 512, 0, 0);
  for (int d = 0; d < 2; ++d)
    gru_scan<<<1, 256, 0, stream>>>(gi[d], whh_t[1][d], bhhp[1][d], gru_out, d * 256, d);

  // pooling: feat0 = (adj3 @ gru_out) / wnt
  wnt_kernel<<<1, 256, 0, stream>>>(adj3, wnt);
  av_gemm<<<dim3(2, 8, 4), 256, 0, stream>>>(adj3, 128, gru_out, 128, wnt,
                                             feat0, 512, 0, 0);

  // 6 relational GAT heads, 2 layers each
  for (int h = 0; h < 6; ++h) {
    for (int l = 0; l < 2; ++l) {
      const float* feat_in = (l == 0) ? feat0 : featw;
      // gate = sigmoid(feat @ fw^T + fb) : [256,768]
      wmma_gemm<<<dim3(8, 12), 256, 0, stream>>>(feat_in, nullptr, nullptr,
          fw_t[h], f_b + (size_t)h * 768, gateb, 256, 768, 512, 1, 0);
      // a = (rel*gate) @ aw^T + ab : [16384,512]  (dominant GEMM, fused A)
      wmma_gemm<<<dim3(512, 8), 256, 0, stream>>>(nullptr, rel_adj, gateb,
          aw_t[h], a_b + (size_t)h * 512, abuf, 16384, 512, 768, 0, 1);
      // c = leaky(LN(a)); logits = c @ dw + db
      ln_leaky_dot<<<2048, 256, 0, stream>>>(abuf, ln_g + (size_t)h * 512,
          ln_b + (size_t)h * 512, d_wp + (size_t)h * 512, d_bp + h, logits);
      // att = softmax(mask(logits)) * adj
      softmax_att<<<32, 256, 0, stream>>>(logits, rel_mask, att);
      // feat' = att @ feat (relu on layer 0; layer 1 writes concat slice)
      if (l == 0)
        av_gemm<<<dim3(2, 8, 4), 256, 0, stream>>>(att, 64, feat_in, 64,
            nullptr, featw, 512, 0, 1);
      else
        av_gemm<<<dim3(2, 8, 4), 256, 0, stream>>>(att, 64, feat_in, 64,
            nullptr, concat, 3072, h * 512, 0);
    }
  }

  // merge: out = concat @ merge_w^T + merge_b : [256,256]
  wmma_gemm<<<dim3(8, 4), 256, 0, stream>>>(concat, nullptr, nullptr,
      merge_t, merge_b, (float*)d_out, 256, 256, 3072, 0, 0);
}
